// LiteMLA_77309411826
// MI455X (gfx1250) — matrix operand order repaired and checked by reading, not compiled, not run on hardware
//
#include <hip/hip_runtime.h>

#define B_      4
#define C_      512
#define H_      64
#define W_      64
#define NPIX    (H_ * W_)          // 4096
#define QKV_CH  1536
#define OUT_CH  1024               // 2*TD

typedef float  v8f   __attribute__((ext_vector_type(8)));
typedef __bf16 v16bf __attribute__((ext_vector_type(16)));

union Frag32B { v16bf v; uint4 u[2]; };

__device__ __forceinline__ unsigned short f2bf(float f) {
  unsigned u = __float_as_uint(f);
  unsigned r = u + 0x7FFFu + ((u >> 16) & 1u);   // round-to-nearest-even
  return (unsigned short)(r >> 16);
}
__device__ __forceinline__ unsigned pack2bf(float lo, float hi) {
  return ((unsigned)f2bf(hi) << 16) | (unsigned)f2bf(lo);
}

#if defined(__HIP_DEVICE_COMPILE__) && \
    __has_builtin(__builtin_amdgcn_global_load_async_to_lds_b32)
#define HAVE_ASYNC_LDS 1
typedef __attribute__((address_space(1))) int* gint_p;
typedef __attribute__((address_space(3))) int* lint_p;
#endif

__device__ __forceinline__ void wait_async0() {
#if __has_builtin(__builtin_amdgcn_s_wait_asynccnt)
  __builtin_amdgcn_s_wait_asynccnt(0);
#elif defined(__HIP_DEVICE_COMPILE__)
  asm volatile("s_wait_asynccnt 0x0" ::: "memory");
#endif
}

// ---------------------------------------------------------------------------
// GEMM: C[b] = A[M,K] * B[b][K,N], bf16 WMMA 16x16x32, f32 accumulate.
// Compile-time shapes -> all address math is shifts.
// Block tile 128x128, K-step 32, 256 threads = 8 waves (2M x 4N), each wave a
// 64x32 tile = 4x2 WMMA accumulators.
// LDS (bf16, row stride 40 shorts = 80B, keeps every fragment read 16B-aligned):
//   As[m][k] (A row-major), Bs[n][k] (B transposed).
// Global loads are float4; LDS stores are packed-bf16 b64/b32.
// ---------------------------------------------------------------------------
template <int M, int N, int K, bool FUSE_BN>
__global__ __launch_bounds__(256) void gemm_wmma_bf16(
    const float* __restrict__ A, const float* __restrict__ Bm,
    float* __restrict__ Cm, size_t strideB, size_t strideC,
    const float* __restrict__ bng, const float* __restrict__ bnb,
    const float* __restrict__ bnm, const float* __restrict__ bnv) {
  const int bz = blockIdx.z;
  const float* __restrict__ Bp = Bm + (size_t)bz * strideB;
  float* __restrict__ Cp = Cm + (size_t)bz * strideC;
  const int m0 = blockIdx.y * 128;
  const int n0 = blockIdx.x * 128;

  __shared__ unsigned short As[128][40];
  __shared__ unsigned short Bs[128][40];

  const int tid   = threadIdx.x;
  const int lane  = tid & 31;
  const int wave  = tid >> 5;
  const int waveM = wave >> 2;   // 0..1
  const int waveN = wave & 3;    // 0..3
  const int half  = lane >> 4;   // 0/1
  const int l16   = lane & 15;

  v8f acc[4][2] = {};

#pragma unroll 2
  for (int k0 = 0; k0 < K; k0 += 32) {
    {  // A tile: 128x32 f32 -> bf16.  float4 loads along K, b64 LDS stores.
      const int c4 = (tid & 7) * 4;     // 0..28
      const int rb = tid >> 3;          // 0..31
#pragma unroll
      for (int i = 0; i < 4; ++i) {
        const int m = rb + i * 32;
        const float4 f = *(const float4*)&A[(size_t)(m0 + m) * K + (k0 + c4)];
        uint2 p;
        p.x = pack2bf(f.x, f.y);
        p.y = pack2bf(f.z, f.w);
        *(uint2*)&As[m][c4] = p;
      }
    }
    {  // B tile transposed: Bs[n][k]=B[k][n]. Two k-rows at once -> b32 stores.
      const int n4 = (tid & 31) * 4;    // 0..124
      const int k2 = (tid >> 5) * 2;    // 0..14
#pragma unroll
      for (int i = 0; i < 2; ++i) {
        const int kk = k2 + i * 16;
        const float4 f0 = *(const float4*)&Bp[(size_t)(k0 + kk) * N + (n0 + n4)];
        const float4 f1 =
            *(const float4*)&Bp[(size_t)(k0 + kk + 1) * N + (n0 + n4)];
        *(unsigned*)&Bs[n4 + 0][kk] = pack2bf(f0.x, f1.x);
        *(unsigned*)&Bs[n4 + 1][kk] = pack2bf(f0.y, f1.y);
        *(unsigned*)&Bs[n4 + 2][kk] = pack2bf(f0.z, f1.z);
        *(unsigned*)&Bs[n4 + 3][kk] = pack2bf(f0.w, f1.w);
      }
    }
    if (k0 + 32 < K) {  // prefetch next K-tile (global_prefetch_b8)
      __builtin_prefetch(
          &A[(size_t)(m0 + (tid >> 3)) * K + (k0 + 32 + (tid & 7) * 4)], 0, 0);
      __builtin_prefetch(
          &Bp[(size_t)(k0 + 32 + (tid >> 5)) * N + (n0 + (tid & 31) * 4)], 0, 0);
    }
    __syncthreads();

    // A frag (16x32 bf16): lane row M=l16; VGPR0-3: K=half*8+{0..7},
    // VGPR4-7: K=16+half*8+{0..7}  -> two aligned ds_load_b128.
    Frag32B af[4], bfr[2];
#pragma unroll
    for (int wm = 0; wm < 4; ++wm) {
      const int row = waveM * 64 + wm * 16 + l16;
      af[wm].u[0] = *(const uint4*)&As[row][half * 8];
      af[wm].u[1] = *(const uint4*)&As[row][16 + half * 8];
    }
    // B frag (32x16 bf16): lane col N=l16; VGPR0-3: K=half*8+{0..7},
    // VGPR4-7: K=16+half*8+{0..7} (same order as A).
#pragma unroll
    for (int wn = 0; wn < 2; ++wn) {
      const int col = waveN * 32 + wn * 16 + l16;
      bfr[wn].u[0] = *(const uint4*)&Bs[col][half * 8];
      bfr[wn].u[1] = *(const uint4*)&Bs[col][16 + half * 8];
    }
#pragma unroll
    for (int wm = 0; wm < 4; ++wm)
#pragma unroll
      for (int wn = 0; wn < 2; ++wn)
        acc[wm][wn] = __builtin_amdgcn_wmma_f32_16x16x32_bf16(
            false, af[wm].v, false, bfr[wn].v, (short)0, acc[wm][wn], false,
            false);
    __syncthreads();
  }

  // D layout: lane col N=l16; VGPR r -> row M = half*8 + r.
#pragma unroll
  for (int wm = 0; wm < 4; ++wm) {
#pragma unroll
    for (int wn = 0; wn < 2; ++wn) {
      const int col = n0 + waveN * 32 + wn * 16 + l16;
#pragma unroll
      for (int r = 0; r < 8; ++r) {
        const int row = m0 + waveM * 64 + wm * 16 + half * 8 + r;
        float val = acc[wm][wn][r];
        if (FUSE_BN) {
          const float inv = bng[row] * rsqrtf(bnv[row] + 1e-5f);
          val = val * inv + (bnb[row] - bnm[row] * inv);
        }
        Cp[(size_t)row * N + col] = val;
      }
    }
  }
}

// ---------------------------------------------------------------------------
// Depthwise 5x5 conv, pad 2. One block = one (batch, channel, 16x16 tile).
// Halo tile staged via async global->LDS copies when available.
// ---------------------------------------------------------------------------
__global__ __launch_bounds__(256) void dwconv5(const float* __restrict__ in,
                                               const float* __restrict__ w,
                                               float* __restrict__ out) {
  const int tileId = blockIdx.x;               // 0..15 (4x4 tiles of 16x16)
  const int ty0 = (tileId >> 2) * 16;
  const int tx0 = (tileId & 3) * 16;
  const int c = blockIdx.y;
  const int b = blockIdx.z;
  const float* __restrict__ ip = in + ((size_t)b * QKV_CH + c) * NPIX;

  __shared__ float tile[20][20];
  __shared__ float wf[25];
  const int tid = threadIdx.x;
  if (tid < 25) wf[tid] = w[c * 25 + tid];

#if defined(HAVE_ASYNC_LDS)
  for (int i = tid; i < 400; i += 256) ((float*)tile)[i] = 0.f;
  __syncthreads();
  for (int i = tid; i < 400; i += 256) {
    const int ly = i / 20, lx = i % 20;
    const int gy = ty0 - 2 + ly, gx = tx0 - 2 + lx;
    if (gy >= 0 && gy < H_ && gx >= 0 && gx < W_) {
      __builtin_amdgcn_global_load_async_to_lds_b32(
          (gint_p)(ip + gy * W_ + gx), (lint_p)&tile[ly][lx], 0, 0);
    }
  }
  wait_async0();
  __syncthreads();
#else
  for (int i = tid; i < 400; i += 256) {
    const int ly = i / 20, lx = i % 20;
    const int gy = ty0 - 2 + ly, gx = tx0 - 2 + lx;
    float v = 0.f;
    if (gy >= 0 && gy < H_ && gx >= 0 && gx < W_) v = ip[gy * W_ + gx];
    tile[ly][lx] = v;
  }
  __syncthreads();
#endif

  const int ly = tid >> 4, lx = tid & 15;
  float s = 0.f;
#pragma unroll
  for (int dy = 0; dy < 5; ++dy)
#pragma unroll
    for (int dx = 0; dx < 5; ++dx) s += tile[ly + dy][lx + dx] * wf[dy * 5 + dx];
  out[((size_t)b * QKV_CH + c) * NPIX + (ty0 + ly) * W_ + (tx0 + lx)] = s;
}

// ---------------------------------------------------------------------------
// Linear attention per (batch, head). heads 0..63 read qkv channels
// [24h..24h+23]; heads 64..127 read dw channels [24h'..24h'+23] and apply the
// three 8x8 w_pw matrices (groups 3h'+{0,1,2}) on the fly.
// Pass1: kv[d][e] = sum_n k[d]*v[e] (v[8]=1). Pass2: out = (q.kv)/(q.kv[:,8]+eps)
// ---------------------------------------------------------------------------
__global__ __launch_bounds__(256) void attn_kernel(
    const float* __restrict__ qkv, const float* __restrict__ dw,
    const float* __restrict__ w_pw, float* __restrict__ attn) {
  const int h = blockIdx.x;   // 0..127
  const int b = blockIdx.y;
  const int tid = threadIdx.x;
  const int lane = tid & 31;

  const bool useAgg = (h >= 64);
  const int hh = useAgg ? (h - 64) : h;
  const float* __restrict__ base =
      (useAgg ? dw : qkv) + ((size_t)b * QKV_CH + 24 * hh) * NPIX;

  __shared__ float kvS[72];        // kv[d][e], e in [0,9)
  __shared__ float pw[3][8][8];    // q/k/v pointwise mats
  if (useAgg) {
    for (int i = tid; i < 192; i += 256) ((float*)pw)[i] = w_pw[3 * hh * 64 + i];
  }
  if (tid < 72) kvS[tid] = 0.f;
  __syncthreads();

  float kvL[8][9];
#pragma unroll
  for (int d = 0; d < 8; ++d)
#pragma unroll
    for (int e = 0; e < 9; ++e) kvL[d][e] = 0.f;

  for (int n = tid; n < NPIX; n += 256) {
    float vals[24];
#pragma unroll
    for (int j = 0; j < 24; ++j) vals[j] = base[(size_t)j * NPIX + n];
    float kd[8], vd[9];
    if (!useAgg) {
#pragma unroll
      for (int d = 0; d < 8; ++d) kd[d] = fmaxf(vals[8 + d], 0.f);
#pragma unroll
      for (int e = 0; e < 8; ++e) vd[e] = vals[16 + e];
    } else {
#pragma unroll
      for (int o = 0; o < 8; ++o) {
        float sk = 0.f, sv = 0.f;
#pragma unroll
        for (int i = 0; i < 8; ++i) {
          sk += pw[1][o][i] * vals[8 + i];
          sv += pw[2][o][i] * vals[16 + i];
        }
        kd[o] = fmaxf(sk, 0.f);
        vd[o] = sv;
      }
    }
    vd[8] = 1.f;
#pragma unroll
    for (int d = 0; d < 8; ++d)
#pragma unroll
      for (int e = 0; e < 9; ++e) kvL[d][e] += kd[d] * vd[e];
  }

  // wave reduce then one LDS atomic per wave per element
#pragma unroll
  for (int d = 0; d < 8; ++d)
#pragma unroll
    for (int e = 0; e < 9; ++e) {
      float v = kvL[d][e];
      for (int off = 16; off > 0; off >>= 1) v += __shfl_down(v, off, 32);
      if (lane == 0) atomicAdd(&kvS[d * 9 + e], v);
    }
  __syncthreads();

  float kv[8][9];
#pragma unroll
  for (int d = 0; d < 8; ++d)
#pragma unroll
    for (int e = 0; e < 9; ++e) kv[d][e] = kvS[d * 9 + e];

  float* __restrict__ outp = attn + ((size_t)b * OUT_CH + 8 * h) * NPIX;
  for (int n = tid; n < NPIX; n += 256) {
    float q[8];
    if (!useAgg) {
#pragma unroll
      for (int d = 0; d < 8; ++d)
        q[d] = fmaxf(base[(size_t)d * NPIX + n], 0.f);
    } else {
      float vals[8];
#pragma unroll
      for (int i = 0; i < 8; ++i) vals[i] = base[(size_t)i * NPIX + n];
#pragma unroll
      for (int o = 0; o < 8; ++o) {
        float s = 0.f;
#pragma unroll
        for (int i = 0; i < 8; ++i) s += pw[0][o][i] * vals[i];
        q[o] = fmaxf(s, 0.f);
      }
    }
    float den = 1e-5f;
#pragma unroll
    for (int d = 0; d < 8; ++d) den += q[d] * kv[d][8];
    const float rden = 1.f / den;
#pragma unroll
    for (int e = 0; e < 8; ++e) {
      float o = 0.f;
#pragma unroll
      for (int d = 0; d < 8; ++d) o += q[d] * kv[d][e];
      outp[(size_t)e * NPIX + n] = o * rden;
    }
  }
}

// ---------------------------------------------------------------------------
extern "C" void kernel_launch(void* const* d_in, const int* in_sizes, int n_in,
                              void* d_out, int out_size, void* d_ws,
                              size_t ws_size, hipStream_t stream) {
  const float* x       = (const float*)d_in[0];
  const float* w_qkv   = (const float*)d_in[1];
  const float* w_dw    = (const float*)d_in[2];
  const float* w_pw    = (const float*)d_in[3];
  const float* w_proj  = (const float*)d_in[4];
  const float* bng     = (const float*)d_in[5];
  const float* bnb     = (const float*)d_in[6];
  const float* bnm     = (const float*)d_in[7];
  const float* bnv     = (const float*)d_in[8];
  float* out = (float*)d_out;

  float* qkv  = (float*)d_ws;                                   // [4,1536,4096]
  float* dw   = qkv + (size_t)B_ * QKV_CH * NPIX;               // [4,1536,4096]
  float* attn = dw + (size_t)B_ * QKV_CH * NPIX;                // [4,1024,4096]

  // 1) qkv = w_qkv[1536,512] @ x[b][512,4096]
  gemm_wmma_bf16<QKV_CH, NPIX, C_, false>
      <<<dim3(NPIX / 128, QKV_CH / 128, B_), 256, 0, stream>>>(
          w_qkv, x, qkv, (size_t)C_ * NPIX, (size_t)QKV_CH * NPIX, bng, bnb,
          bnm, bnv);

  // 2) depthwise 5x5
  dwconv5<<<dim3(16, QKV_CH, B_), 256, 0, stream>>>(qkv, w_dw, dw);

  // 3) pointwise 8x8 + linear attention + normalize -> attn
  attn_kernel<<<dim3(128, B_), 256, 0, stream>>>(qkv, dw, w_pw, attn);

  // 4) y = w_proj[512,1024] @ attn[b][1024,4096], fused BN
  gemm_wmma_bf16<C_, NPIX, OUT_CH, true>
      <<<dim3(NPIX / 128, C_ / 128, B_), 256, 0, stream>>>(
          w_proj, attn, out, (size_t)OUT_CH * NPIX, (size_t)C_ * NPIX, bng,
          bnb, bnm, bnv);
}
